// BaselineModel_34608846471244
// MI455X (gfx1250) — matrix operand order repaired and checked by reading, not compiled, not run on hardware
//
#include <hip/hip_runtime.h>
#include <hip/hip_bf16.h>
#include <math.h>

typedef __attribute__((ext_vector_type(16))) _Float16 v16h;
typedef __attribute__((ext_vector_type(8)))  _Float16 v8h;
typedef __attribute__((ext_vector_type(4)))  _Float16 v4h;
typedef __attribute__((ext_vector_type(8)))  float    v8f;

#define B_   8
#define T_   2048
#define D_   128
#define L_   4
#define H_   2
#define HD_  64
#define MT_  (B_*T_)
#define DD_  (D_*D_)
#define EPS_ 1e-8f

__device__ __forceinline__ v8f wmma_f16(v16h a, v16h b, v8f c) {
  return __builtin_amdgcn_wmma_f32_16x16x32_f16(false, a, false, b, (short)0, c, false, false);
}
#define CAT16(a0, a1) __builtin_shufflevector(a0, a1, 0,1,2,3,4,5,6,7,8,9,10,11,12,13,14,15)

__device__ __forceinline__ float fast_rcp(float x)  { return __builtin_amdgcn_rcpf(x); }
__device__ __forceinline__ float fast_rsq(float x)  { return __builtin_amdgcn_rsqf(x); }
__device__ __forceinline__ float fast_sigmoid(float v) {
  return fast_rcp(1.f + __expf(-v));
}

// ---------------- weight f32 -> f16 ----------------
__global__ void cvt_f16(const float* __restrict__ s, _Float16* __restrict__ d, int n) {
  int i = blockIdx.x * blockDim.x + threadIdx.x;
  if (i < n) d[i] = (_Float16)s[i];
}

// ---------------- embedding gather + RMSNorm ----------------
__global__ __launch_bounds__(256) void embed_rms(const int* __restrict__ ids,
    const float* __restrict__ itab, const float* __restrict__ ptab,
    const float* __restrict__ sc, float* __restrict__ x) {
  int tid = threadIdx.x, lane = tid & 31, wv = tid >> 5;
  size_t tok = (size_t)blockIdx.x * 8 + wv;
  int t = (int)(tok % T_);
  int id = ids[tok];
  const float4 e = *(const float4*)(itab + (size_t)id * D_ + lane * 4);
  const float4 p = *(const float4*)(ptab + (size_t)(t + 1) * D_ + lane * 4);
  float4 v; v.x = e.x + p.x; v.y = e.y + p.y; v.z = e.z + p.z; v.w = e.w + p.w;
  float ss = v.x * v.x + v.y * v.y + v.z * v.z + v.w * v.w;
  #pragma unroll
  for (int m = 1; m < 32; m <<= 1) ss += __shfl_xor(ss, m, 32);
  float inv = fast_rsq(ss * (1.f / D_) + EPS_);
  const float4 s4 = *(const float4*)(sc + lane * 4);
  float4 o; o.x = v.x * inv * s4.x; o.y = v.y * inv * s4.y;
  o.z = v.z * inv * s4.z; o.w = v.w * inv * s4.w;
  *(float4*)(x + tok * D_ + lane * 4) = o;
}

// ---------------- RMSNorm f32 -> f16 ----------------
__global__ __launch_bounds__(256) void rms_f16(const float* __restrict__ src,
    const float* __restrict__ sc, _Float16* __restrict__ dst) {
  int tid = threadIdx.x, lane = tid & 31, wv = tid >> 5;
  size_t tok = (size_t)blockIdx.x * 8 + wv;
  const float4 v = *(const float4*)(src + tok * D_ + lane * 4);
  float ss = v.x * v.x + v.y * v.y + v.z * v.z + v.w * v.w;
  #pragma unroll
  for (int m = 1; m < 32; m <<= 1) ss += __shfl_xor(ss, m, 32);
  float inv = fast_rsq(ss * (1.f / D_) + EPS_);
  const float4 s4 = *(const float4*)(sc + lane * 4);
  v4h o; o[0] = (_Float16)(v.x * inv * s4.x); o[1] = (_Float16)(v.y * inv * s4.y);
  o[2] = (_Float16)(v.z * inv * s4.z); o[3] = (_Float16)(v.w * inv * s4.w);
  *(v4h*)(dst + tok * D_ + lane * 4) = o;
}

// ---------------- HSTU mix: RMS(AV)*U -> f16 ----------------
__global__ __launch_bounds__(256) void hstu_mix(const float* __restrict__ AV,
    const float* __restrict__ sc, const float* __restrict__ U, _Float16* __restrict__ y) {
  int tid = threadIdx.x, lane = tid & 31, wv = tid >> 5;
  size_t tok = (size_t)blockIdx.x * 8 + wv;
  const float4 v = *(const float4*)(AV + tok * D_ + lane * 4);
  float ss = v.x * v.x + v.y * v.y + v.z * v.z + v.w * v.w;
  #pragma unroll
  for (int m = 1; m < 32; m <<= 1) ss += __shfl_xor(ss, m, 32);
  float inv = fast_rsq(ss * (1.f / D_) + EPS_);
  const float4 s4 = *(const float4*)(sc + lane * 4);
  const float4 u  = *(const float4*)(U + tok * D_ + lane * 4);
  v4h o; o[0] = (_Float16)(v.x * inv * s4.x * u.x); o[1] = (_Float16)(v.y * inv * s4.y * u.y);
  o[2] = (_Float16)(v.z * inv * s4.z * u.z); o[3] = (_Float16)(v.w * inv * s4.w * u.w);
  *(v4h*)(y + tok * D_ + lane * 4) = o;
}

// ---------------- final RMSNorm f32 -> f32 ----------------
__global__ __launch_bounds__(256) void final_rms(const float* __restrict__ src,
    const float* __restrict__ sc, float* __restrict__ dst) {
  int tid = threadIdx.x, lane = tid & 31, wv = tid >> 5;
  size_t tok = (size_t)blockIdx.x * 8 + wv;
  const float4 v = *(const float4*)(src + tok * D_ + lane * 4);
  float ss = v.x * v.x + v.y * v.y + v.z * v.z + v.w * v.w;
  #pragma unroll
  for (int m = 1; m < 32; m <<= 1) ss += __shfl_xor(ss, m, 32);
  float inv = fast_rsq(ss * (1.f / D_) + EPS_);
  const float4 s4 = *(const float4*)(sc + lane * 4);
  float4 o; o.x = v.x * inv * s4.x; o.y = v.y * inv * s4.y;
  o.z = v.z * inv * s4.z; o.w = v.w * inv * s4.w;
  *(float4*)(dst + tok * D_ + lane * 4) = o;
}

// ---------------- V (b,t,D) f16 -> Vt (b, c, t) f16, 32x32 LDS tiles ----------------
__global__ __launch_bounds__(256) void transpose_v(const _Float16* __restrict__ V,
                                                   _Float16* __restrict__ Vt) {
  __shared__ _Float16 tile[32][40];
  int b = blockIdx.z;
  int c0 = blockIdx.y * 32;
  int s0 = blockIdx.x * 32;
  int tx = threadIdx.x & 31, ty = threadIdx.x >> 5;
  #pragma unroll
  for (int i = 0; i < 4; ++i) {
    int s = s0 + ty + i * 8;
    tile[ty + i * 8][tx] = V[((size_t)b * T_ + s) * D_ + c0 + tx];
  }
  __syncthreads();
  #pragma unroll
  for (int i = 0; i < 4; ++i) {
    int c = c0 + ty + i * 8;
    Vt[((size_t)b * D_ + c) * T_ + s0 + tx] = tile[tx][ty + i * 8];
  }
}

// ---------------- 16384x128x128 WMMA GEMM, weight resident in LDS ----------------
// ACT: 0=none 1=silu 2=gelu(exact). RESID: out = resid + gemm. OUT32/OUT16 select outputs.
template<int ACT, bool RESID, bool OUT32, bool OUT16>
__global__ __launch_bounds__(256) void gemm128(const _Float16* __restrict__ A,
    const _Float16* __restrict__ W, const float* __restrict__ bias,
    float* __restrict__ outF, _Float16* __restrict__ outH) {
  __shared__ alignas(32) _Float16 ldsW[D_ * D_];   // stored transposed: [n][k]
  int tid = threadIdx.x;
  #pragma unroll
  for (int i = 0; i < 8; ++i) {
    int e8 = tid + i * 256;            // chunk of 8 halves
    int k = e8 >> 4;                   // (e8*8)/128
    int n = (e8 & 15) * 8;             // (e8*8)%128
    v8h w = *(const v8h*)(W + k * D_ + n);
    #pragma unroll
    for (int j = 0; j < 8; ++j) ldsW[(n + j) * D_ + k] = w[j];
  }
  __syncthreads();

  int lane = tid & 31, wv = tid >> 5;
  int l16 = lane & 15, hi = lane >> 4;
  int row0 = blockIdx.x * 128 + wv * 16;

  v8f zero;
  #pragma unroll
  for (int j = 0; j < 8; ++j) zero[j] = 0.f;
  v8f acc[8];
  #pragma unroll
  for (int nt = 0; nt < 8; ++nt) acc[nt] = zero;

  const _Float16* arow = A + (size_t)(row0 + l16) * D_;
  #pragma unroll
  for (int ks = 0; ks < 4; ++ks) {
    const _Float16* ap = arow + ks * 32 + hi * 16;
    v8h a0 = *(const v8h*)ap;
    v8h a1 = *(const v8h*)(ap + 8);
    v16h af = CAT16(a0, a1);
    #pragma unroll
    for (int nt = 0; nt < 8; ++nt) {
      v16h bf = *(const v16h*)(&ldsW[(nt * 16 + l16) * D_ + ks * 32 + hi * 16]);
      acc[nt] = wmma_f16(af, bf, acc[nt]);
    }
  }

  #pragma unroll
  for (int nt = 0; nt < 8; ++nt) {
    int col = nt * 16 + l16;
    float bv = bias[col];
    #pragma unroll
    for (int r = 0; r < 8; ++r) {
      int row = row0 + r + 8 * hi;
      size_t idx = (size_t)row * D_ + col;
      float v = acc[nt][r] + bv;
      if (ACT == 1) v = v * fast_sigmoid(v);
      else if (ACT == 2) v = 0.5f * v * (1.f + erff(v * 0.70710678118654752f));
      if (RESID) v += outF[idx];
      if (OUT32) outF[idx] = v;
      if (OUT16) outH[idx] = (_Float16)v;
    }
  }
}

// ---------------- HSTU causal attention (flash-style, unnormalized accumulate) ----------------
__global__ __launch_bounds__(256) void hstu_attn(const _Float16* __restrict__ Q,
    const _Float16* __restrict__ K, const _Float16* __restrict__ Vt,
    float* __restrict__ AV) {
  __shared__ alignas(16) _Float16 ldsA[8][16 * 40];   // per-wave 16x32 A-tile, stride 40
  int tid = threadIdx.x, lane = tid & 31, wv = tid >> 5;
  int l16 = lane & 15, hi = lane >> 4;
  int h = blockIdx.y, b = blockIdx.z;
  int r0 = (blockIdx.x * 8 + wv) * 16;

  const size_t base = ((size_t)b * T_) * D_ + h * HD_;
  const _Float16* Qp = Q + base;
  const _Float16* Kp = K + base;
  const _Float16* Vtb = Vt + ((size_t)b * D_ + h * HD_) * T_;   // [d][s]
  _Float16* As = ldsA[wv];

  // hoisted Q fragments (prescaled by 1/sqrt(HD), exact in f16): 16 rows x 64 features
  v16h qf[2];
  #pragma unroll
  for (int ks = 0; ks < 2; ++ks) {
    const _Float16* ap = Qp + (size_t)(r0 + l16) * D_ + ks * 32 + hi * 16;
    v8h a0 = *(const v8h*)ap;
    v8h a1 = *(const v8h*)(ap + 8);
    qf[ks] = CAT16(a0, a1) * (_Float16)0.125f;
  }

  v8f o[4];
  #pragma unroll
  for (int nt = 0; nt < 4; ++nt)
    #pragma unroll
    for (int j = 0; j < 8; ++j) o[nt][j] = 0.f;
  float rs[8];
  #pragma unroll
  for (int r = 0; r < 8; ++r) rs[r] = 0.f;

  for (int c0 = 0; c0 < r0 + 16; c0 += 32) {
    // two 16-col score tiles -> masked SiLU (branch-free) -> LDS (A-layout source)
    #pragma unroll
    for (int t = 0; t < 2; ++t) {
      int colb = c0 + t * 16 + l16;       // provably < T_
      v8f s;
      #pragma unroll
      for (int j = 0; j < 8; ++j) s[j] = 0.f;
      #pragma unroll
      for (int ks = 0; ks < 2; ++ks) {
        v16h bf = *(const v16h*)(Kp + (size_t)colb * D_ + ks * 32 + hi * 16);
        s = wmma_f16(qf[ks], bf, s);
      }
      #pragma unroll
      for (int r = 0; r < 8; ++r) {
        int row = r0 + r + 8 * hi;
        float v = s[r];
        float num = (colb <= row) ? fmaxf(v, 0.f) : 0.f;  // mask * relu(v)
        float a = num * fast_sigmoid(v);                  // max(silu(v),0), masked
        rs[r] += a;
        As[(r + 8 * hi) * 40 + t * 16 + l16] = (_Float16)a;
      }
    }
    asm volatile("" ::: "memory");   // keep ds_store before ds_load (HW keeps in-wave LDS order)

    // A fragment (16 rows x 32 keys) from LDS
    const _Float16* arp = As + l16 * 40 + hi * 16;
    v8h a0 = *(const v8h*)arp;
    v8h a1 = *(const v8h*)(arp + 8);
    v16h af = CAT16(a0, a1);

    // accumulate AV over 4 output n-tiles; B-frag is one contiguous v16h from Vt
    #pragma unroll
    for (int nt = 0; nt < 4; ++nt) {
      v16h bf = *(const v16h*)(Vtb + (size_t)(nt * 16 + l16) * T_ + c0 + hi * 16);
      o[nt] = wmma_f16(af, bf, o[nt]);
    }
  }

  // reduce row sums across the 16 column-lanes of each half
  #pragma unroll
  for (int r = 0; r < 8; ++r) {
    float v = rs[r];
    v += __shfl_xor(v, 1, 32);
    v += __shfl_xor(v, 2, 32);
    v += __shfl_xor(v, 4, 32);
    v += __shfl_xor(v, 8, 32);
    rs[r] = (v > 1e-12f) ? fast_rcp(v + EPS_) : 0.f;
  }

  #pragma unroll
  for (int nt = 0; nt < 4; ++nt)
    #pragma unroll
    for (int r = 0; r < 8; ++r) {
      size_t idx = ((size_t)b * T_ + r0 + r + 8 * hi) * D_ + h * HD_ + nt * 16 + l16;
      AV[idx] = o[nt][r] * rs[r];
    }
}

// ---------------- host launch ----------------
extern "C" void kernel_launch(void* const* d_in, const int* in_sizes, int n_in,
                              void* d_out, int out_size, void* d_ws, size_t ws_size,
                              hipStream_t stream) {
  (void)in_sizes; (void)n_in; (void)out_size; (void)ws_size;
  const int*   ids   = (const int*)d_in[0];
  // d_in[1] = attn_mask (fixed causal tril; handled analytically in hstu_attn)
  const float* itab  = (const float*)d_in[2];
  const float* ptab  = (const float*)d_in[3];
  const float* emb_s = (const float*)d_in[4];
  const float* ln1_s = (const float*)d_in[5];
  const float* Uw = (const float*)d_in[6];  const float* Ubias = (const float*)d_in[7];
  const float* Vw = (const float*)d_in[8];  const float* Vbias = (const float*)d_in[9];
  const float* Qw = (const float*)d_in[10]; const float* Qbias = (const float*)d_in[11];
  const float* Kw = (const float*)d_in[12]; const float* Kbias = (const float*)d_in[13];
  const float* F2w = (const float*)d_in[14]; const float* F2b = (const float*)d_in[15];
  const float* hstu_s = (const float*)d_in[16];
  const float* ln2_s  = (const float*)d_in[17];
  const float* C1w = (const float*)d_in[18]; const float* C1b = (const float*)d_in[19];
  const float* C2w = (const float*)d_in[20]; const float* C2b = (const float*)d_in[21];
  const float* last_s = (const float*)d_in[22];
  float* out = (float*)d_out;

  char* p = (char*)d_ws;
  auto take = [&](size_t bytes) -> char* {
    char* r = p; p += (bytes + 255) & ~(size_t)255; return r;
  };
  float*    x    = (float*)take((size_t)MT_ * D_ * 4);
  _Float16* xn   = (_Float16*)take((size_t)MT_ * D_ * 2);
  float*    Uact = (float*)take((size_t)MT_ * D_ * 4);
  _Float16* Qh   = (_Float16*)take((size_t)MT_ * D_ * 2);
  _Float16* Kh   = (_Float16*)take((size_t)MT_ * D_ * 2);
  _Float16* Vh   = (_Float16*)take((size_t)MT_ * D_ * 2);
  _Float16* Vt   = (_Float16*)take((size_t)MT_ * D_ * 2 + 256);  // [b][c][t], padded
  float*    AV   = (float*)take((size_t)MT_ * D_ * 4);
  _Float16* y16  = (_Float16*)take((size_t)MT_ * D_ * 2);
  _Float16* w16[7];
  for (int i = 0; i < 7; ++i) w16[i] = (_Float16*)take((size_t)L_ * DD_ * 2);

  const float* wsrc[7] = {Uw, Vw, Qw, Kw, F2w, C1w, C2w};
  for (int i = 0; i < 7; ++i)
    cvt_f16<<<(L_ * DD_ + 255) / 256, 256, 0, stream>>>(wsrc[i], w16[i], L_ * DD_);

  dim3 tb(256);
  embed_rms<<<MT_ / 8, tb, 0, stream>>>(ids, itab, ptab, emb_s, x);

  for (int l = 0; l < L_; ++l) {
    size_t wo = (size_t)l * DD_;
    rms_f16<<<MT_ / 8, tb, 0, stream>>>(x, ln1_s + l * D_, xn);
    gemm128<1, false, true,  false><<<MT_ / 128, tb, 0, stream>>>(xn, w16[0] + wo, Ubias + l * D_, Uact, nullptr);
    gemm128<1, false, false, true ><<<MT_ / 128, tb, 0, stream>>>(xn, w16[1] + wo, Vbias + l * D_, nullptr, Vh);
    gemm128<1, false, false, true ><<<MT_ / 128, tb, 0, stream>>>(xn, w16[2] + wo, Qbias + l * D_, nullptr, Qh);
    gemm128<1, false, false, true ><<<MT_ / 128, tb, 0, stream>>>(xn, w16[3] + wo, Kbias + l * D_, nullptr, Kh);
    transpose_v<<<dim3(T_ / 32, D_ / 32, B_), tb, 0, stream>>>(Vh, Vt);
    hstu_attn<<<dim3(T_ / 16 / 8, H_, B_), tb, 0, stream>>>(Qh, Kh, Vt, AV);
    hstu_mix<<<MT_ / 8, tb, 0, stream>>>(AV, hstu_s + l * D_, Uact, y16);
    gemm128<0, true,  true,  false><<<MT_ / 128, tb, 0, stream>>>(y16, w16[4] + wo, F2b + l * D_, x, nullptr);
    rms_f16<<<MT_ / 8, tb, 0, stream>>>(x, ln2_s + l * D_, xn);
    gemm128<2, false, false, true ><<<MT_ / 128, tb, 0, stream>>>(xn, w16[5] + wo, C1b + l * D_, nullptr, y16);
    gemm128<0, true,  true,  false><<<MT_ / 128, tb, 0, stream>>>(y16, w16[6] + wo, C2b + l * D_, x, nullptr);
  }
  final_rms<<<MT_ / 8, tb, 0, stream>>>(x, last_s, out);
}
